// LSTM_63574105916135
// MI455X (gfx1250) — compile-verified, matrix operand order checked by reading, not run on hardware
//
#include <hip/hip_runtime.h>
#include <hip/hip_bf16.h>

// ---------------------------------------------------------------------------
// LSTM scan for MI455X (gfx1250, wave32, WMMA).
//   Phase 1: xz[s,b,h] = sum_i x[b,s,i]*Wx[h,i] + bias[h]   (bf16 WMMA GEMM)
//   Phase 2: persistent scan kernel, 1024 steps, grid barrier per step.
// Workspace layout (needs ~133 MB):
//   [0,2MB)       Wh bf16  [1024][1024]
//   [2MB,4MB)     Wx bf16  [1024][1024]
//   [4MB,+128KB)  h double buffer bf16 [2][32][1024]
//   [4MB+256KB)   barrier counters (2 x u32)
//   [5MB,+128MB)  xz f32 [S*B][H]
// ---------------------------------------------------------------------------

typedef __bf16 bf16_t;
typedef __attribute__((ext_vector_type(16))) __bf16 v16bf;
typedef __attribute__((ext_vector_type(8)))  __bf16 v8bf;
typedef __attribute__((ext_vector_type(8)))  float  v8f;

#define B_  32
#define S_  1024
#define I_  1024
#define H_  1024

union FragU { v16bf v; v8bf h[2]; };

// 16x32 bf16 WMMA operand fragment (A-layout, ISA 7.12.2):
// lane l: row = l&15, hs = l>>4;
//   elems 0..7  = K[hs*8 .. +7], elems 8..15 = K[16+hs*8 .. +7]
// -> exactly two contiguous 16-byte loads per lane.
__device__ __forceinline__ v16bf frag_from(const bf16_t* p0, const bf16_t* p1) {
    FragU u;
    u.h[0] = *(const v8bf*)p0;
    u.h[1] = *(const v8bf*)p1;
    return u.v;
}

// Build an A fragment directly from a per-lane f32 row (converted to bf16).
__device__ __forceinline__ v16bf frag_from_f32(const float* row_k) {
    const float4* p0 = (const float4*)(row_k);        // K = base..base+7
    const float4* p1 = (const float4*)(row_k + 16);   // K = base+16..+23
    float4 f0 = p0[0], f1 = p0[1], g0 = p1[0], g1 = p1[1];
    v16bf a;
    a[0]=(bf16_t)f0.x; a[1]=(bf16_t)f0.y; a[2]=(bf16_t)f0.z; a[3]=(bf16_t)f0.w;
    a[4]=(bf16_t)f1.x; a[5]=(bf16_t)f1.y; a[6]=(bf16_t)f1.z; a[7]=(bf16_t)f1.w;
    a[8]=(bf16_t)g0.x; a[9]=(bf16_t)g0.y; a[10]=(bf16_t)g0.z; a[11]=(bf16_t)g0.w;
    a[12]=(bf16_t)g1.x; a[13]=(bf16_t)g1.y; a[14]=(bf16_t)g1.z; a[15]=(bf16_t)g1.w;
    return a;
}

// ---------------------------------------------------------------------------
// Kernel 1: split/convert W (f32, H x (H+I)) -> Wh bf16, Wx bf16 (row-major)
// ---------------------------------------------------------------------------
__global__ void convert_w(const float* __restrict__ W,
                          bf16_t* __restrict__ Wh,
                          bf16_t* __restrict__ Wx) {
    int id = blockIdx.x * blockDim.x + threadIdx.x;   // H*(I+H) = 2M threads
    if (id >= H_ * (I_ + H_)) return;
    int h = id >> 11;          // / 2048
    int c = id & 2047;
    float v = W[id];
    if (c < H_) Wh[(size_t)h * H_ + c]        = (bf16_t)v;
    else        Wx[(size_t)h * I_ + (c - H_)] = (bf16_t)v;
}

__global__ void convert_h0(const float* __restrict__ h0,
                           bf16_t* __restrict__ hbuf0) {
    int id = blockIdx.x * blockDim.x + threadIdx.x;   // B*H = 32768
    if (id < B_ * H_) hbuf0[id] = (bf16_t)h0[id];
}

// ---------------------------------------------------------------------------
// Kernel 2: xz GEMM.  M = S*B = 32768 (row m = s*32+b), N = H, K = I.
// Workgroup = 256 threads (8 waves), tile 128(M) x 128(N), K-step 32.
// A: loaded per-lane straight from global f32 (x is streamed once, only
//    intra-wave reuse -> no LDS roundtrip; full 128B-line use per row/slab).
// B: Wx bf16 tile staged in double-buffered LDS (8x cross-wave reuse),
//    pitch 40 halves -> conflict-free b128 fragment reads, ONE barrier/iter.
// All 8 B fragments preloaded before the 8-WMMA chain (one dscnt wait).
// ---------------------------------------------------------------------------
__global__ __launch_bounds__(256) void xz_gemm(const float*  __restrict__ x,
                                               const bf16_t* __restrict__ Wx,
                                               const float*  __restrict__ bias,
                                               float*        __restrict__ xz) {
    __shared__ __align__(16) bf16_t Bbuf[2][128][40];

    const int nblk = blockIdx.x & 7;    // 8 N blocks
    const int mblk = blockIdx.x >> 3;   // 256 M blocks
    const int m0 = mblk * 128;
    const int n0 = nblk * 128;

    const int t    = threadIdx.x;
    const int lane = t & 31;
    const int wave = t >> 5;            // 0..7
    const int ls   = lane & 15;
    const int hs   = lane >> 4;

    const v8f zero = {0.f,0.f,0.f,0.f,0.f,0.f,0.f,0.f};
    v8f acc[8];
#pragma unroll
    for (int i = 0; i < 8; ++i) acc[i] = zero;

    // Per-lane A row: m = m0 + wave*16 + ls -> (s,b) -> f32 row of x.
    const int m = m0 + wave * 16 + ls;
    const int sidx = m >> 5, bidx = m & 31;
    const float* arow = x + ((size_t)bidx * S_ + sidx) * I_;

    // B staging: thread t covers row rr = t>>1, 16-col half hsel = t&1.
    const int rr   = t >> 1;
    const int hsel = t & 1;
    const bf16_t* bsrc_row = Wx + (size_t)(n0 + rr) * I_ + hsel * 16;

    // Prologue: stage first B tile into buffer 0.
    {
        const uint4* src = (const uint4*)(bsrc_row);
        uint4* dst = (uint4*)&Bbuf[0][rr][hsel * 16];
        dst[0] = src[0]; dst[1] = src[1];
    }
    __syncthreads();

    for (int ki = 0; ki < I_ / 32; ++ki) {
        const int cur = ki & 1;
        // Stage next B tile into the other buffer (overlaps with compute).
        if (ki + 1 < I_ / 32) {
            const uint4* src = (const uint4*)(bsrc_row + (ki + 1) * 32);
            uint4* dst = (uint4*)&Bbuf[cur ^ 1][rr][hsel * 16];
            dst[0] = src[0]; dst[1] = src[1];
        }
        // A fragment straight from global (f32 -> bf16 in regs).
        v16bf a = frag_from_f32(arow + ki * 32 + hs * 8);
        // Preload all 8 B fragments, then the WMMA chain.
        v16bf bb[8];
#pragma unroll
        for (int nt = 0; nt < 8; ++nt)
            bb[nt] = frag_from(&Bbuf[cur][nt * 16 + ls][hs * 8],
                               &Bbuf[cur][nt * 16 + ls][16 + hs * 8]);
#pragma unroll
        for (int nt = 0; nt < 8; ++nt)
            acc[nt] = __builtin_amdgcn_wmma_f32_16x16x32_bf16(
                          false, a, false, bb[nt], (short)0, acc[nt], false, false);
        __syncthreads();
    }

    // Epilogue: + bias, store f32.  Acc elem r: M = hs*8 + r, N = ls.
    const int mrow = m0 + wave * 16 + hs * 8;
#pragma unroll
    for (int nt = 0; nt < 8; ++nt) {
        int n = n0 + nt * 16 + ls;
        float bv = bias[n];
#pragma unroll
        for (int r = 0; r < 8; ++r)
            xz[(size_t)(mrow + r) * H_ + n] = acc[nt][r] + bv;
    }
}

// ---------------------------------------------------------------------------
// Kernel 3: persistent scan.  128 single-wave (32-thread) workgroups.
// wg = (mt, nt): mt in {0,1} -> 16 rows of B, nt in 0..63 -> 16 cols of H.
// Wh tile (16 x 1024 bf16 = 32KB) staged in LDS once; c state in registers;
// h passed between steps through a bf16 double buffer in global (L2-resident).
// K-loop software-pipelined one iteration ahead so fragment loads overlap the
// serial WMMA accumulation chain.  One grid barrier per step.
// ---------------------------------------------------------------------------
__global__ __launch_bounds__(32) void lstm_scan(const float*  __restrict__ xz,
                                                const bf16_t* __restrict__ Whb,
                                                bf16_t*       __restrict__ hbuf,   // [2][32][1024]
                                                const float*  __restrict__ c0,
                                                float*        __restrict__ out,    // out | hn | cn
                                                unsigned*     __restrict__ bar,
                                                int nwg) {
    __shared__ __align__(16) bf16_t WhT[16][1032];   // pitch 1032 halves: conflict-free

    const int wg = blockIdx.x;          // 0..127
    const int mt = wg & 1;
    const int nt = wg >> 1;
    const int m0 = mt * 16;
    const int n0 = nt * 16;
    const int lane = threadIdx.x;
    const int ls = lane & 15;
    const int hs = lane >> 4;

    // Preload Wh rows n0..n0+15 (B operand, stable across all 1024 steps).
    for (int i = lane; i < 16 * 128; i += 32) {      // 128 x 16B chunks per row
        int row = i >> 7;
        int c8  = (i & 127) * 8;
        *(uint4*)&WhT[row][c8] = *(const uint4*)(Whb + (size_t)(n0 + row) * H_ + c8);
    }
    __syncthreads();

    // c state in accumulator layout: elem r -> (b = m0+hs*8+r, h = n0+ls)
    v8f c;
#pragma unroll
    for (int r = 0; r < 8; ++r)
        c[r] = c0[(size_t)(m0 + hs * 8 + r) * H_ + n0 + ls];

    float hlast[8];

    for (int s = 0; s < S_; ++s) {
        const int p = s & 1;
        const bf16_t* hsrc = hbuf + (size_t)p * (B_ * H_);
        bf16_t*       hdst = hbuf + (size_t)(p ^ 1) * (B_ * H_);

        // z initialized from xz[s] (row m = s*32 + b)
        v8f z;
        const float* xzp = xz + ((size_t)s * B_ + m0 + hs * 8) * H_ + n0 + ls;
#pragma unroll
        for (int r = 0; r < 8; ++r) z[r] = xzp[(size_t)r * H_];

        // Prefetch next step's xz tile (global_prefetch_b8).
        if (s + 1 < S_) {
            const float* nxt = xz + ((size_t)(s + 1) * B_ + m0 + hs * 8) * H_ + n0 + ls;
            __builtin_prefetch(nxt, 0, 1);
        }

        // z += h @ Wh^T : 32 chained v_wmma_f32_16x16x32_bf16, loads pipelined
        // one K-slab ahead of the accumulator chain.
        const bf16_t* hrow = hsrc + (size_t)(m0 + ls) * H_ + hs * 8;
        v16bf a  = frag_from(hrow, hrow + 16);
        v16bf bb = frag_from(&WhT[ls][hs * 8], &WhT[ls][16 + hs * 8]);
#pragma unroll 4
        for (int kk = 32; kk < H_; kk += 32) {
            v16bf a2 = frag_from(hrow + kk, hrow + kk + 16);
            v16bf b2 = frag_from(&WhT[ls][kk + hs * 8], &WhT[ls][kk + 16 + hs * 8]);
            z = __builtin_amdgcn_wmma_f32_16x16x32_bf16(
                    false, a, false, bb, (short)0, z, false, false);
            a = a2; bb = b2;
        }
        z = __builtin_amdgcn_wmma_f32_16x16x32_bf16(
                false, a, false, bb, (short)0, z, false, false);

        // Elementwise gate math + stores.
#pragma unroll
        for (int r = 0; r < 8; ++r) {
            float zv  = z[r];
            float sg  = 1.0f / (1.0f + __expf(-zv));
            float thz = 2.0f / (1.0f + __expf(-2.0f * zv)) - 1.0f;   // tanh(z)
            float cn  = sg * c[r] + sg * thz;
            c[r] = cn;
            float thc = 2.0f / (1.0f + __expf(-2.0f * cn)) - 1.0f;   // tanh(c_new)
            float hn  = sg * thc;
            hlast[r] = hn;
            int b = m0 + hs * 8 + r;
            // out[b][s][h] — streamed once, bypass cache pollution
            __builtin_nontemporal_store(hn,
                &out[(size_t)b * (S_ * H_) + (size_t)s * H_ + n0 + ls]);
            hdst[(size_t)b * H_ + n0 + ls] = (bf16_t)hn;             // L2-resident
        }

        // --- grid barrier (double-buffered h => one barrier per step) ---
        __builtin_amdgcn_fence(__ATOMIC_RELEASE, "agent");
        if (lane == 0) {
            unsigned old = __hip_atomic_fetch_add(&bar[0], 1u, __ATOMIC_ACQ_REL,
                                                  __HIP_MEMORY_SCOPE_AGENT);
            if (old == (unsigned)nwg - 1u) {
                __hip_atomic_store(&bar[0], 0u, __ATOMIC_RELAXED,
                                   __HIP_MEMORY_SCOPE_AGENT);
                __hip_atomic_fetch_add(&bar[1], 1u, __ATOMIC_RELEASE,
                                       __HIP_MEMORY_SCOPE_AGENT);
            }
        }
        const unsigned target = (unsigned)(s + 1);
        while (__hip_atomic_load(&bar[1], __ATOMIC_ACQUIRE,
                                 __HIP_MEMORY_SCOPE_AGENT) < target) {
            __builtin_amdgcn_s_sleep(2);
        }
        __builtin_amdgcn_fence(__ATOMIC_ACQUIRE, "agent");
    }

    // hn, cn outputs (concatenated after out)
    float* hn_out = out + (size_t)B_ * S_ * H_;
    float* cn_out = hn_out + B_ * H_;
#pragma unroll
    for (int r = 0; r < 8; ++r) {
        int b = m0 + hs * 8 + r;
        hn_out[(size_t)b * H_ + n0 + ls] = hlast[r];
        cn_out[(size_t)b * H_ + n0 + ls] = c[r];
    }
}

// ---------------------------------------------------------------------------
extern "C" void kernel_launch(void* const* d_in, const int* in_sizes, int n_in,
                              void* d_out, int out_size, void* d_ws, size_t ws_size,
                              hipStream_t stream) {
    (void)in_sizes; (void)n_in; (void)out_size; (void)ws_size;

    const float* x    = (const float*)d_in[0];   // (B,S,I)
    const float* h0   = (const float*)d_in[1];   // (1,B,H)
    const float* c0   = (const float*)d_in[2];   // (1,B,H)
    const float* W    = (const float*)d_in[3];   // (H, H+I)
    const float* bias = (const float*)d_in[4];   // (H,)
    float* out = (float*)d_out;

    char* ws = (char*)d_ws;
    bf16_t*   Whb  = (bf16_t*)(ws);
    bf16_t*   Wxb  = (bf16_t*)(ws + (size_t)2 * 1024 * 1024);
    bf16_t*   hbuf = (bf16_t*)(ws + (size_t)4 * 1024 * 1024);
    unsigned* bar  = (unsigned*)(ws + (size_t)4 * 1024 * 1024 + 256 * 1024);
    float*    xz   = (float*)(ws + (size_t)5 * 1024 * 1024);

    (void)hipMemsetAsync(bar, 0, 256, stream);

    convert_w <<<(H_ * (I_ + H_)) / 256, 256, 0, stream>>>(W, Whb, Wxb);
    convert_h0<<<(B_ * H_) / 256,        256, 0, stream>>>(h0, hbuf);
    xz_gemm   <<<2048, 256, 0, stream>>>(x, Wxb, bias, xz);
    lstm_scan <<<128,   32, 0, stream>>>(xz, Whb, hbuf, c0, out, bar, 128);
}
